// CRDLoss_83210696393682
// MI455X (gfx1250) — compile-verified, matrix operand order checked by reading, not run on hardware
//
#include <hip/hip_runtime.h>
#include <hip/hip_bf16.h>

typedef __attribute__((ext_vector_type(16))) __bf16 v16bf;
typedef __attribute__((ext_vector_type(8)))  __bf16 v8bf;
typedef __attribute__((ext_vector_type(8)))  float  v8f;

#define T_INV     14.2857142857142857f   // 1 / 0.07
#define EPS_C     0.97f
#define LOG_EPS  (-0.030459207484708574f) // log(0.97)

// ---- fragment loaders -------------------------------------------------------
// A-matrix 16x32 bf16 layout (ISA 7.12.2): lane L<16 -> row M=L, K = k0..k0+7
// (VGPR0..3) and k0+16..k0+23 (VGPR4..7); lanes 16..31 -> K spans +8.
// B-matrix 32x16 is the mirror (lane holds column N), so the same loader works
// for B when 'row' is the column-tile base of a row-major [N][K] matrix.

__device__ __forceinline__ v16bf load_frag_bf16(const __bf16* __restrict__ base,
                                                int row, int ld, int k0, int lane) {
    const int r    = row + (lane & 15);
    const int koff = k0 + ((lane >> 4) << 3);
    const __bf16* p = base + (size_t)r * ld + koff;
    v8bf lo = *(const v8bf*)(p);
    v8bf hi = *(const v8bf*)(p + 16);
    v16bf out;
#pragma unroll
    for (int i = 0; i < 8; ++i) { out[i] = lo[i]; out[8 + i] = hi[i]; }
    return out;
}

__device__ __forceinline__ v16bf load_frag_f32(const float* __restrict__ base,
                                               int row, int ld, int k0, int lane) {
    const int r    = row + (lane & 15);
    const int koff = k0 + ((lane >> 4) << 3);
    const float* p = base + (size_t)r * ld + koff;
    v16bf out;
#pragma unroll
    for (int i = 0; i < 8; ++i) out[i]     = (__bf16)p[i];
#pragma unroll
    for (int i = 0; i < 8; ++i) out[8 + i] = (__bf16)p[16 + i];
    return out;
}

// ---- kernel 1: fused Linear + bias + L2-normalize -> bf16 embeddings --------
// Block = 256 threads (8 waves). Block computes a 16-row x 128-col tile.
// Wave w owns column tile [w*16, w*16+16). blockIdx.y selects student/teacher.
__global__ __launch_bounds__(256) void crd_embed_kernel(
    const float* __restrict__ Xs, const float* __restrict__ Ws, const float* __restrict__ bs,
    const float* __restrict__ Xt, const float* __restrict__ Wt, const float* __restrict__ bt,
    __bf16* __restrict__ outS, __bf16* __restrict__ outT, int Ks, int Kt)
{
    const float* X; const float* W; const float* bvp; __bf16* out; int K;
    if (blockIdx.y == 0) { X = Xs; W = Ws; bvp = bs; out = outS; K = Ks; }
    else                 { X = Xt; W = Wt; bvp = bt; out = outT; K = Kt; }

    const int rowBase = blockIdx.x * 16;
    const int lane = threadIdx.x & 31;
    const int wave = threadIdx.x >> 5;
    const int c0   = wave * 16;

    v8f acc = {};
    for (int k = 0; k < K; k += 32) {
        v16bf a = load_frag_f32(X, rowBase, K, k, lane);   // activations tile
        v16bf b = load_frag_f32(W, c0,      K, k, lane);   // weight rows = output cols
        acc = __builtin_amdgcn_wmma_f32_16x16x32_bf16(false, a, false, b,
                                                      (short)0, acc, false, false);
    }

    __shared__ float hbuf[16][132];
    __shared__ float rinv[16];

    const float bv   = bvp[c0 + (lane & 15)];
    const int  mbase = (lane >> 4) << 3;
    const int  col   = c0 + (lane & 15);
#pragma unroll
    for (int i = 0; i < 8; ++i) hbuf[mbase + i][col] = acc[i] + bv;
    __syncthreads();

    if (threadIdx.x < 16) {
        float s = 0.f;
#pragma unroll 4
        for (int c = 0; c < 128; ++c) { float v = hbuf[threadIdx.x][c]; s += v * v; }
        rinv[threadIdx.x] = rsqrtf(s);
    }
    __syncthreads();

    const int idx = threadIdx.x * 8;        // 2048 outputs / 256 threads
    const int r = idx >> 7, c = idx & 127;
    const float iv = rinv[r];
    __bf16* dst = out + (size_t)(rowBase + r) * 128 + c;
#pragma unroll
    for (int i = 0; i < 8; ++i) dst[i] = (__bf16)(hbuf[r][c + i] * iv);
}

// ---- kernel 2: fused similarity GEMM + contrastive-loss epilogue ------------
// Block = 256 threads (8 waves) computes a 128x128 tile of fs @ ft^T.
// The shared ft tile (32 KB) is staged into LDS ONCE per block with CDNA5
// async copies (GLOBAL_LOAD_ASYNC_TO_LDS_B128, tracked on ASYNCcnt), then all
// 8 waves feed their B fragments from LDS (conflict-free via 136-elem stride).
// Wave w owns rows [w*16, w*16+16) x all 128 cols: 8 wmma accumulators.

#define FT_LDS_STRIDE 136   // bf16 elements per LDS row (272 B = 16B-aligned, pads banks)

__global__ __launch_bounds__(256) void crd_sim_loss_kernel(
    const __bf16* __restrict__ fs, const __bf16* __restrict__ ft,
    const int* __restrict__ y, float* __restrict__ partials, int nbx)
{
    const int rowBase = blockIdx.y * 128;
    const int colBase = blockIdx.x * 128;
    const int lane = threadIdx.x & 31;
    const int wave = threadIdx.x >> 5;

    __shared__ __bf16 ftTile[128 * FT_LDS_STRIDE];
    __shared__ int yrow[128];
    __shared__ int ycol[128];

    // ---- async stage of the 128x128 ft tile: 2048 x 16B chunks, 8 per thread
    {
        const unsigned ldsBase = (unsigned)(uintptr_t)(void*)&ftTile[0];
        const __bf16* gBase = ft + (size_t)colBase * 128;
#pragma unroll
        for (int it = 0; it < 8; ++it) {
            const int c   = threadIdx.x + it * 256;  // chunk id 0..2047
            const int row = c >> 4;                  // 16 chunks (256B) per row
            const int cc  = c & 15;
            const unsigned ldsOff = ldsBase + (unsigned)(row * (FT_LDS_STRIDE * 2) + cc * 16);
            const unsigned long long gaddr =
                (unsigned long long)(uintptr_t)(gBase + (size_t)row * 128 + cc * 8);
            asm volatile("global_load_async_to_lds_b128 %0, %1, off"
                         :: "v"(ldsOff), "v"(gaddr) : "memory");
        }
    }
    if (threadIdx.x < 128) yrow[threadIdx.x]       = y[rowBase + threadIdx.x];
    else                   ycol[threadIdx.x - 128] = y[colBase + threadIdx.x - 128];
    asm volatile("s_wait_asynccnt 0x0" ::: "memory");
    __syncthreads();

    const int wrow = rowBase + wave * 16;
    v8f acc[8];
#pragma unroll
    for (int n = 0; n < 8; ++n) acc[n] = (v8f){};

#pragma unroll
    for (int k = 0; k < 128; k += 32) {
        v16bf a = load_frag_bf16(fs, wrow, 128, k, lane);  // own rows: direct global
        const int koff = k + ((lane >> 4) << 3);
#pragma unroll
        for (int n = 0; n < 8; ++n) {
            // B fragment from LDS: lane holds column N = n*16 + (lane&15)
            const __bf16* p = &ftTile[(n * 16 + (lane & 15)) * FT_LDS_STRIDE + koff];
            v8bf lo = *(const v8bf*)(p);        // ds_load_b128
            v8bf hi = *(const v8bf*)(p + 16);   // ds_load_b128
            v16bf b;
#pragma unroll
            for (int i = 0; i < 8; ++i) { b[i] = lo[i]; b[8 + i] = hi[i]; }
            acc[n] = __builtin_amdgcn_wmma_f32_16x16x32_bf16(false, a, false, b,
                                                             (short)0, acc[n], false, false);
        }
    }

    // Epilogue: term = diag ? 0 : (same ? s/T : log(eps)) - log(exp(s/T)+eps)
    const int mbase = (lane >> 4) << 3;
    const int nlane = lane & 15;
    float sum = 0.f;
#pragma unroll
    for (int n = 0; n < 8; ++n) {
        const int gcolL = n * 16 + nlane;
        const int gcol  = colBase + gcolL;
        const int yc    = ycol[gcolL];
#pragma unroll
        for (int i = 0; i < 8; ++i) {
            const int rl   = wave * 16 + mbase + i;
            const int grow = rowBase + rl;
            const float t   = acc[n][i] * T_INV;
            const float x   = __expf(t);
            const float lxe = __logf(x + EPS_C);
            const bool diag = (grow == gcol);
            const bool same = (yrow[rl] == yc);
            float term = (same ? t : LOG_EPS) - lxe;
            sum += diag ? 0.0f : term;
        }
    }

    // wave32 tree reduce, then cross-wave in LDS
#pragma unroll
    for (int off = 16; off > 0; off >>= 1) sum += __shfl_down(sum, off, 32);
    __shared__ float wsum[8];
    if (lane == 0) wsum[wave] = sum;
    __syncthreads();
    if (threadIdx.x == 0) {
        float t = 0.f;
#pragma unroll
        for (int w = 0; w < 8; ++w) t += wsum[w];
        partials[blockIdx.y * nbx + blockIdx.x] = t;
    }
}

// ---- kernel 3: deterministic final reduction --------------------------------
__global__ __launch_bounds__(256) void crd_reduce_kernel(
    const float* __restrict__ partials, int n, float* __restrict__ out, float negInvB)
{
    float s = 0.f;
    for (int i = threadIdx.x; i < n; i += 256) s += partials[i];
#pragma unroll
    for (int off = 16; off > 0; off >>= 1) s += __shfl_down(s, off, 32);
    __shared__ float ws[8];
    if ((threadIdx.x & 31) == 0) ws[threadIdx.x >> 5] = s;
    __syncthreads();
    if (threadIdx.x == 0) {
        float t = 0.f;
#pragma unroll
        for (int w = 0; w < 8; ++w) t += ws[w];
        out[0] = t * negInvB;   // loss = -(sum)/B
    }
}

extern "C" void kernel_launch(void* const* d_in, const int* in_sizes, int n_in,
                              void* d_out, int out_size, void* d_ws, size_t ws_size,
                              hipStream_t stream) {
    const float* f_s = (const float*)d_in[0];
    const float* f_t = (const float*)d_in[1];
    const int*   y   = (const int*)  d_in[2];
    const float* W_s = (const float*)d_in[3];
    const float* b_s = (const float*)d_in[4];
    const float* W_t = (const float*)d_in[5];
    const float* b_t = (const float*)d_in[6];

    const int B  = in_sizes[2];             // 8192
    const int Ks = in_sizes[0] / B;         // 2048
    const int Kt = in_sizes[1] / B;         // 2048
    (void)n_in; (void)out_size; (void)ws_size;

    // workspace layout: fs_bf16 [B*128], ft_bf16 [B*128], partials [(B/128)^2]
    __bf16* fsb = (__bf16*)d_ws;
    __bf16* ftb = fsb + (size_t)B * 128;
    float*  part = (float*)(ftb + (size_t)B * 128);

    dim3 g1(B / 16, 2);
    crd_embed_kernel<<<g1, 256, 0, stream>>>(f_s, W_s, b_s, f_t, W_t, b_t,
                                             fsb, ftb, Ks, Kt);

    const int nbx = B / 128;
    dim3 g2(nbx, nbx);
    crd_sim_loss_kernel<<<g2, 256, 0, stream>>>(fsb, ftb, y, part, nbx);

    crd_reduce_kernel<<<1, 256, 0, stream>>>(part, nbx * nbx, (float*)d_out,
                                             -1.0f / (float)B);
}
